// DEEPGCN_27212912788336
// MI455X (gfx1250) — matrix operand order repaired and checked by reading, not compiled, not run on hardware
//
#include <hip/hip_runtime.h>

typedef __attribute__((ext_vector_type(16))) _Float16 v16h;
typedef __attribute__((ext_vector_type(8)))  float    v8f;

#define HID 64
#define INF 128
#define OUTF 32
#define EPS_MSG 1e-7f
#define EPS_BN 1e-5f

static __device__ __forceinline__ v8f wmma_f16(v16h a, v16h b, v8f c) {
  return __builtin_amdgcn_wmma_f32_16x16x32_f16(false, a, false, b, (short)0, c, false, false);
}

// A-fragment K index for element e (0..15), lane-high bit hi (0/1):
//   K = e + hi*16
#define KOFS(e, hi) ((e) + (hi) * 16)

// ---------------------------------------------------------------- init
__global__ void k_init(float* zmax, float* denom, float* num, float* stats, int total) {
  int i = blockIdx.x * blockDim.x + threadIdx.x;
  if (i < total) { zmax[i] = 0.0f; denom[i] = 0.0f; num[i] = 0.0f; }
  if (blockIdx.x == 0 && threadIdx.x < 256) stats[threadIdx.x] = 0.0f;
}

// ------------------------------------------------- h = x @ Wsrc^T + bsrc
// block = 256 thr = 8 waves; block covers 32 rows x 64 cols; wave -> 16x16 tile
__global__ void k_gemm_src(const float* __restrict__ x, const float* __restrict__ Wsrc,
                           const float* __restrict__ bsrc, float* __restrict__ h, int N) {
  int lane = threadIdx.x & 31, wave = threadIdx.x >> 5;
  int m16 = lane & 15, hi = lane >> 4;
  int r0 = blockIdx.x * 32 + (wave >> 2) * 16;
  int c0 = (wave & 3) * 16;
  int arow = min(r0 + m16, N - 1);
  v8f acc = {};
  for (int k0 = 0; k0 < INF; k0 += 32) {
    v16h a, b;
    const float* xr = x + (long long)arow * INF + k0;
    const float* wr = Wsrc + (c0 + m16) * INF + k0 + hi * 16;
#pragma unroll
    for (int e = 0; e < 16; ++e) {
      a[e] = (_Float16)xr[KOFS(e, hi)];
      b[e] = (_Float16)wr[e];
    }
    acc = wmma_f16(a, b, acc);
  }
  int n = c0 + m16;
  float bb = bsrc[n];
#pragma unroll
  for (int r = 0; r < 8; ++r) {
    int row = r0 + r + hi * 8;
    if (row < N) h[(long long)row * HID + n] = acc[r] + bb;
  }
}

// ------------------------------------------- edge pass 1: segment max (int trick)
__global__ void k_edges1(const int* __restrict__ ei, const float* __restrict__ h,
                         float* __restrict__ zmax, long long total, int E) {
  long long idx = (long long)blockIdx.x * blockDim.x + threadIdx.x;
  if (idx >= total) return;
  int edge = (int)(idx >> 6), ch = (int)(idx & 63);
  int s = ei[edge], d = ei[E + edge];
  float msg = fmaxf(h[(long long)s * HID + ch], 0.0f) + EPS_MSG;  // always > 0
  atomicMax((int*)(zmax + (long long)d * HID + ch), __float_as_int(msg));
}

// ------------------------------------------- edge pass 2: exp-weighted sums
__global__ void k_edges2(const int* __restrict__ ei, const float* __restrict__ h,
                         const float* __restrict__ zmax, float* __restrict__ denom,
                         float* __restrict__ num, long long total, int E) {
  long long idx = (long long)blockIdx.x * blockDim.x + threadIdx.x;
  if (idx >= total) return;
  int edge = (int)(idx >> 6), ch = (int)(idx & 63);
  int s = ei[edge], d = ei[E + edge];
  float msg = fmaxf(h[(long long)s * HID + ch], 0.0f) + EPS_MSG;
  long long o = (long long)d * HID + ch;
  float ez = __expf(msg - zmax[o]);
  atomicAdd(denom + o, ez);
  atomicAdd(num + o, ez * msg);
}

// -------------------------- y1 = (agg + h) @ W1^T + b1   (16 rows x 128 cols/block)
__global__ void k_mlp1(const float* __restrict__ h, const float* __restrict__ num,
                       const float* __restrict__ denom, const float* __restrict__ W1,
                       const float* __restrict__ b1, float* __restrict__ y1, int N) {
  __shared__ _Float16 lds[16 * HID];
  int r0 = blockIdx.x * 16;
  for (int i = threadIdx.x; i < 16 * HID; i += blockDim.x) {
    int row = i >> 6, ch = i & 63;
    long long o = (long long)min(r0 + row, N - 1) * HID + ch;
    float agg = num[o] / fmaxf(denom[o], 1e-16f);
    lds[i] = (_Float16)(agg + h[o]);
  }
  __syncthreads();
  int lane = threadIdx.x & 31, wave = threadIdx.x >> 5;
  int m16 = lane & 15, hi = lane >> 4;
  int c0 = wave * 16;
  v8f acc = {};
  for (int k0 = 0; k0 < HID; k0 += 32) {
    v16h a, b;
    const float* wr = W1 + (c0 + m16) * HID + k0 + hi * 16;
#pragma unroll
    for (int e = 0; e < 16; ++e) {
      a[e] = lds[m16 * HID + k0 + KOFS(e, hi)];
      b[e] = (_Float16)wr[e];
    }
    acc = wmma_f16(a, b, acc);
  }
  int n = c0 + m16;
  float bb = b1[n];
#pragma unroll
  for (int r = 0; r < 8; ++r) {
    int row = r0 + r + hi * 8;
    if (row < N) y1[(long long)row * INF + n] = acc[r] + bb;
  }
}

// -------------------------------- BN stats: per-channel sum / sumsq over N rows
__global__ void k_bnstats(const float* __restrict__ y1, float* __restrict__ stats,
                          long long total) {
  __shared__ float s1[256], s2[256];
  float a = 0.0f, b = 0.0f;
  long long stride = (long long)gridDim.x * blockDim.x;  // multiple of 128
  for (long long i = (long long)blockIdx.x * blockDim.x + threadIdx.x; i < total; i += stride) {
    float v = y1[i];
    a += v; b += v * v;
  }
  s1[threadIdx.x] = a; s2[threadIdx.x] = b;
  __syncthreads();
  if (threadIdx.x < 128) {  // channel == threadIdx.x
    atomicAdd(stats + threadIdx.x,       s1[threadIdx.x] + s1[threadIdx.x + 128]);
    atomicAdd(stats + 128 + threadIdx.x, s2[threadIdx.x] + s2[threadIdx.x + 128]);
  }
}

__global__ void k_bnfinal(const float* __restrict__ stats, const float* __restrict__ gamma,
                          const float* __restrict__ beta, float* __restrict__ bnst, float invN) {
  int c = threadIdx.x;  // 128 threads
  float mu = stats[c] * invN;
  float var = stats[128 + c] * invN - mu * mu;
  float s = gamma[c] * rsqrtf(var + EPS_BN);
  bnst[c] = s;
  bnst[128 + c] = beta[c] - mu * s;
}

// ---- fused: relu(BN(y1)) @ W2^T + b2 -> relu -> @ Wlin^T + blin  (32 rows/block)
__global__ void k_mlp2(const float* __restrict__ y1, const float* __restrict__ bnst,
                       const float* __restrict__ W2, const float* __restrict__ b2,
                       const float* __restrict__ Wlin, const float* __restrict__ blin,
                       float* __restrict__ out, int N) {
  __shared__ _Float16 z[32 * INF];   // 8 KB
  __shared__ _Float16 y2[32 * HID];  // 4 KB
  int r0 = blockIdx.x * 32;
  for (int i = threadIdx.x; i < 32 * INF; i += blockDim.x) {
    int row = i >> 7, ch = i & 127;
    float v = y1[(long long)min(r0 + row, N - 1) * INF + ch];
    z[i] = (_Float16)fmaxf(v * bnst[ch] + bnst[128 + ch], 0.0f);
  }
  __syncthreads();
  int lane = threadIdx.x & 31, wave = threadIdx.x >> 5;
  int m16 = lane & 15, hi = lane >> 4;
  {  // stage 1: 32x128 @ 128x64 -> 8 tiles, one per wave
    int rt = (wave >> 2) * 16, c0 = (wave & 3) * 16;
    v8f acc = {};
    for (int k0 = 0; k0 < INF; k0 += 32) {
      v16h a, b;
      const float* wr = W2 + (c0 + m16) * INF + k0 + hi * 16;
#pragma unroll
      for (int e = 0; e < 16; ++e) {
        a[e] = z[(rt + m16) * INF + k0 + KOFS(e, hi)];
        b[e] = (_Float16)wr[e];
      }
      acc = wmma_f16(a, b, acc);
    }
    int n = c0 + m16;
    float bb = b2[n];
#pragma unroll
    for (int r = 0; r < 8; ++r)
      y2[(rt + r + hi * 8) * HID + n] = (_Float16)fmaxf(acc[r] + bb, 0.0f);
  }
  __syncthreads();
  if (wave < 4) {  // stage 2: 32x64 @ 64x32 -> 4 tiles (wave-uniform branch, EXEC all-1s)
    int rt = (wave >> 1) * 16, c0 = (wave & 1) * 16;
    v8f acc = {};
    for (int k0 = 0; k0 < HID; k0 += 32) {
      v16h a, b;
      const float* wr = Wlin + (c0 + m16) * HID + k0 + hi * 16;
#pragma unroll
      for (int e = 0; e < 16; ++e) {
        a[e] = y2[(rt + m16) * HID + k0 + KOFS(e, hi)];
        b[e] = (_Float16)wr[e];
      }
      acc = wmma_f16(a, b, acc);
    }
    int n = c0 + m16;
    float bb = blin[n];
#pragma unroll
    for (int r = 0; r < 8; ++r) {
      int row = r0 + rt + r + hi * 8;
      if (row < N) out[(long long)row * OUTF + n] = acc[r] + bb;
    }
  }
}

extern "C" void kernel_launch(void* const* d_in, const int* in_sizes, int n_in,
                              void* d_out, int out_size, void* d_ws, size_t ws_size,
                              hipStream_t stream) {
  const float* x    = (const float*)d_in[0];
  const int*   ei   = (const int*)d_in[1];
  const float* Wsrc = (const float*)d_in[2];
  const float* bsrc = (const float*)d_in[3];
  const float* W1   = (const float*)d_in[4];
  const float* b1   = (const float*)d_in[5];
  const float* gamma= (const float*)d_in[6];
  const float* beta = (const float*)d_in[7];
  const float* W2   = (const float*)d_in[8];
  const float* b2   = (const float*)d_in[9];
  const float* Wlin = (const float*)d_in[10];
  const float* blin = (const float*)d_in[11];
  float* outp = (float*)d_out;

  const int N = in_sizes[0] / INF;
  const int E = in_sizes[1] / 2;

  float* ws = (float*)d_ws;
  const long long NH = (long long)N * HID;
  float* h     = ws;
  float* zmax  = ws + NH;
  float* denom = ws + 2 * NH;
  float* num   = ws + 3 * NH;
  float* y1    = ws + 4 * NH;          // N*128 = 2*NH floats
  float* stats = ws + 6 * NH;          // 256 floats
  float* bnst  = ws + 6 * NH + 256;    // 256 floats

  // init zmax/denom/num/stats
  {
    int total = (int)NH;
    k_init<<<(total + 255) / 256, 256, 0, stream>>>(zmax, denom, num, stats, total);
  }
  // h = x @ Wsrc^T + bsrc
  k_gemm_src<<<(N + 31) / 32, 256, 0, stream>>>(x, Wsrc, bsrc, h, N);
  // edge passes
  {
    long long total = (long long)E * HID;
    unsigned blocks = (unsigned)((total + 255) / 256);
    k_edges1<<<blocks, 256, 0, stream>>>(ei, h, zmax, total, E);
    k_edges2<<<blocks, 256, 0, stream>>>(ei, h, zmax, denom, num, total, E);
  }
  // y1 = (agg + h) @ W1^T + b1
  k_mlp1<<<(N + 15) / 16, 256, 0, stream>>>(h, num, denom, W1, b1, y1, N);
  // BN statistics + fold
  k_bnstats<<<1024, 256, 0, stream>>>(y1, stats, (long long)N * INF);
  k_bnfinal<<<1, 128, 0, stream>>>(stats, gamma, beta, bnst, 1.0f / (float)N);
  // fused BN->relu->W2->relu->Wlin
  k_mlp2<<<(N + 31) / 32, 256, 0, stream>>>(y1, bnst, W2, b2, Wlin, blin, outp, N);
}
